// BidirectionalMambaBlock_47545287966870
// MI455X (gfx1250) — compile-verified, matrix operand order checked by reading, not run on hardware
//
#include <hip/hip_runtime.h>
#include <hip/hip_bf16.h>

typedef __attribute__((ext_vector_type(8)))  _Float16 v8h;
typedef __attribute__((ext_vector_type(16))) _Float16 v16h;
typedef __attribute__((ext_vector_type(8)))  float    v8f;

#define B_SZ 2
#define L_SZ 1024
#define DM   512
#define DI   1024
#define DST  16
#define DTR  32
#define ROWS (B_SZ * L_SZ)   // 2048
#define NCH  16              // scan chunks
#define CL   (L_SZ / NCH)    // 64 steps per chunk

// ---------------------------------------------------------------------------
// Fragment loader: lane holds 16 f16 = {K: kb..kb+7} ++ {K: kb+16..kb+23}
// matching the ISA 16-bit A/B 16x32 layout (lanes 0-15 kb=k0, 16-31 kb=k0+8).
// ---------------------------------------------------------------------------
__device__ __forceinline__ v16h load_frag16(const _Float16* __restrict__ p) {
  v8h lo = *(const v8h*)(p);
  v8h hi = *(const v8h*)(p + 16);
  return __builtin_shufflevector(lo, hi, 0,1,2,3,4,5,6,7,8,9,10,11,12,13,14,15);
}

// ---------------------------------------------------------------------------
// WMMA GEMM: C[M,N] (f32) = A[M,K] (f16 row-major) * W[N,K]^T (f16 row-major)
// One wave per block; wave computes a 32x64 output tile (2 A-frags x 4 W-frags
// -> 8 v_wmma per 32-wide K step), with explicit register double-buffering so
// next-iteration loads are in flight while current WMMAs execute.
// grid = (N/64, M/32), block = 32.
// ---------------------------------------------------------------------------
__global__ void gemm_wmma_f32(const _Float16* __restrict__ A,
                              const _Float16* __restrict__ W,
                              float* __restrict__ C,
                              int M, int N, int K)
{
  const int lane = threadIdx.x;
  const int n0   = blockIdx.x * 64;
  const int m0   = blockIdx.y * 32;
  const int mrow = m0 + (lane & 15);
  const int koff = (lane >> 4) * 8;

  v8f acc[8] = {};
  const _Float16* Arow0 = A + (size_t)mrow * K + koff;
  const _Float16* Arow1 = A + (size_t)(mrow + 16) * K + koff;
  const _Float16* Wrow  = W + (size_t)(n0 + (lane & 15)) * K + koff;

  __builtin_prefetch(Arow0, 0, 3);   // WGP-scope: pull into all cache levels
  __builtin_prefetch(Arow1, 0, 3);

  // prologue: fragments for k = 0
  v16h a0 = load_frag16(Arow0);
  v16h a1 = load_frag16(Arow1);
  v16h w0 = load_frag16(Wrow);
  v16h w1 = load_frag16(Wrow + (size_t)16 * K);
  v16h w2 = load_frag16(Wrow + (size_t)32 * K);
  v16h w3 = load_frag16(Wrow + (size_t)48 * K);

  for (int k0 = 0; k0 < K; k0 += 32) {
    // issue next-iteration loads first (clamped to 0 on the last iteration so
    // the loop stays branch-free; the fetched data is never consumed)
    const int kn = (k0 + 32 < K) ? (k0 + 32) : 0;
    v16h na0 = load_frag16(Arow0 + kn);
    v16h na1 = load_frag16(Arow1 + kn);
    v16h nw0 = load_frag16(Wrow + kn);
    v16h nw1 = load_frag16(Wrow + (size_t)16 * K + kn);
    v16h nw2 = load_frag16(Wrow + (size_t)32 * K + kn);
    v16h nw3 = load_frag16(Wrow + (size_t)48 * K + kn);

    acc[0] = __builtin_amdgcn_wmma_f32_16x16x32_f16(false, a0, false, w0, (short)0, acc[0], false, false);
    acc[4] = __builtin_amdgcn_wmma_f32_16x16x32_f16(false, a1, false, w0, (short)0, acc[4], false, false);
    acc[1] = __builtin_amdgcn_wmma_f32_16x16x32_f16(false, a0, false, w1, (short)0, acc[1], false, false);
    acc[5] = __builtin_amdgcn_wmma_f32_16x16x32_f16(false, a1, false, w1, (short)0, acc[5], false, false);
    acc[2] = __builtin_amdgcn_wmma_f32_16x16x32_f16(false, a0, false, w2, (short)0, acc[2], false, false);
    acc[6] = __builtin_amdgcn_wmma_f32_16x16x32_f16(false, a1, false, w2, (short)0, acc[6], false, false);
    acc[3] = __builtin_amdgcn_wmma_f32_16x16x32_f16(false, a0, false, w3, (short)0, acc[3], false, false);
    acc[7] = __builtin_amdgcn_wmma_f32_16x16x32_f16(false, a1, false, w3, (short)0, acc[7], false, false);

    a0 = na0; a1 = na1; w0 = nw0; w1 = nw1; w2 = nw2; w3 = nw3;
  }

  // C/D layout: lane(0-15)->M=r, lane(16-31)->M=8+r ; N = lane&15
  const int nn = lane & 15;
#pragma unroll
  for (int s = 0; s < 2; ++s) {
    const int mb = m0 + s * 16 + (lane >> 4) * 8;
#pragma unroll
    for (int j = 0; j < 4; ++j)
#pragma unroll
      for (int r = 0; r < 8; ++r)
        C[(size_t)(mb + r) * N + (n0 + 16 * j + nn)] = acc[4 * s + j][r];
  }
}

// ---------------------------------------------------------------------------
// Same GEMM but stores f16 into a wider fused buffer, optionally un-flipping
// the time axis (for the backward direction) and with a column offset.
// ---------------------------------------------------------------------------
__global__ void gemm_wmma_f16out(const _Float16* __restrict__ A,
                                 const _Float16* __restrict__ W,
                                 _Float16* __restrict__ Co,
                                 int M, int N, int K,
                                 int outStride, int colOff, int flip)
{
  const int lane = threadIdx.x;
  const int n0   = blockIdx.x * 64;
  const int m0   = blockIdx.y * 32;
  const int mrow = m0 + (lane & 15);
  const int koff = (lane >> 4) * 8;

  v8f acc[8] = {};
  const _Float16* Arow0 = A + (size_t)mrow * K + koff;
  const _Float16* Arow1 = A + (size_t)(mrow + 16) * K + koff;
  const _Float16* Wrow  = W + (size_t)(n0 + (lane & 15)) * K + koff;

  __builtin_prefetch(Arow0, 0, 3);
  __builtin_prefetch(Arow1, 0, 3);

  v16h a0 = load_frag16(Arow0);
  v16h a1 = load_frag16(Arow1);
  v16h w0 = load_frag16(Wrow);
  v16h w1 = load_frag16(Wrow + (size_t)16 * K);
  v16h w2 = load_frag16(Wrow + (size_t)32 * K);
  v16h w3 = load_frag16(Wrow + (size_t)48 * K);

  for (int k0 = 0; k0 < K; k0 += 32) {
    const int kn = (k0 + 32 < K) ? (k0 + 32) : 0;
    v16h na0 = load_frag16(Arow0 + kn);
    v16h na1 = load_frag16(Arow1 + kn);
    v16h nw0 = load_frag16(Wrow + kn);
    v16h nw1 = load_frag16(Wrow + (size_t)16 * K + kn);
    v16h nw2 = load_frag16(Wrow + (size_t)32 * K + kn);
    v16h nw3 = load_frag16(Wrow + (size_t)48 * K + kn);

    acc[0] = __builtin_amdgcn_wmma_f32_16x16x32_f16(false, a0, false, w0, (short)0, acc[0], false, false);
    acc[4] = __builtin_amdgcn_wmma_f32_16x16x32_f16(false, a1, false, w0, (short)0, acc[4], false, false);
    acc[1] = __builtin_amdgcn_wmma_f32_16x16x32_f16(false, a0, false, w1, (short)0, acc[1], false, false);
    acc[5] = __builtin_amdgcn_wmma_f32_16x16x32_f16(false, a1, false, w1, (short)0, acc[5], false, false);
    acc[2] = __builtin_amdgcn_wmma_f32_16x16x32_f16(false, a0, false, w2, (short)0, acc[2], false, false);
    acc[6] = __builtin_amdgcn_wmma_f32_16x16x32_f16(false, a1, false, w2, (short)0, acc[6], false, false);
    acc[3] = __builtin_amdgcn_wmma_f32_16x16x32_f16(false, a0, false, w3, (short)0, acc[3], false, false);
    acc[7] = __builtin_amdgcn_wmma_f32_16x16x32_f16(false, a1, false, w3, (short)0, acc[7], false, false);

    a0 = na0; a1 = na1; w0 = nw0; w1 = nw1; w2 = nw2; w3 = nw3;
  }

  const int nn = lane & 15;
#pragma unroll
  for (int s = 0; s < 2; ++s) {
    const int mb = m0 + s * 16 + (lane >> 4) * 8;
#pragma unroll
    for (int j = 0; j < 4; ++j)
#pragma unroll
      for (int r = 0; r < 8; ++r) {
        int m  = mb + r;
        int ro = m;
        if (flip) {
          int b = m / L_SZ, t = m % L_SZ;
          ro = b * L_SZ + (L_SZ - 1 - t);
        }
        Co[(size_t)ro * outStride + colOff + n0 + 16 * j + nn] =
            (_Float16)acc[4 * s + j][r];
      }
  }
}

// ---------------------------------------------------------------------------
// Elementwise converts
// ---------------------------------------------------------------------------
__global__ void cvt_f16(const float* __restrict__ in, _Float16* __restrict__ out, int n) {
  int i = blockIdx.x * blockDim.x + threadIdx.x;
  if (i < n) out[i] = (_Float16)in[i];
}

__global__ void cvt_f16_flip(const float* __restrict__ in, _Float16* __restrict__ out, int n) {
  int i = blockIdx.x * blockDim.x + threadIdx.x;
  if (i >= n) return;
  int c = i % DM;
  int r = i / DM;
  int b = r / L_SZ, t = r % L_SZ;
  out[(size_t)(b * L_SZ + (L_SZ - 1 - t)) * DM + c] = (_Float16)in[i];
}

__global__ void slice_cvt_dt(const float* __restrict__ xdbl, _Float16* __restrict__ dt16, int n) {
  int i = blockIdx.x * blockDim.x + threadIdx.x;
  if (i >= n) return;
  int r = i >> 5;
  int c = i & 31;
  dt16[i] = (_Float16)xdbl[(size_t)r * 64 + c];
}

// ---------------------------------------------------------------------------
// Causal depthwise conv (width 4) + bias + SiLU on the xi half of xz.
// ---------------------------------------------------------------------------
__global__ void conv_silu(const float* __restrict__ xz,
                          const float* __restrict__ cw,
                          const float* __restrict__ cb,
                          float* __restrict__ xi32,
                          _Float16* __restrict__ xi16)
{
  int i = blockIdx.x * blockDim.x + threadIdx.x;
  if (i >= ROWS * DI) return;
  int d = i & (DI - 1);
  int r = i >> 10;
  int t = r % L_SZ;
  int b = r / L_SZ;
  float acc = cb[d];
#pragma unroll
  for (int j = 0; j < 4; ++j) {
    int tt = t - 3 + j;
    if (tt >= 0)
      acc += xz[(size_t)(b * L_SZ + tt) * (2 * DI) + d] * cw[d * 4 + j];
  }
  float s = acc / (1.0f + __expf(-acc));
  xi32[i] = s;
  xi16[i] = (_Float16)s;
}

// ---------------------------------------------------------------------------
// Chunk-parallel selective scan (diagonal linear recurrence):
// pass1 local scans + per-chunk transition products, pass2 sequential chunk
// stitch (16 steps), pass3 h0 correction + D skip + SiLU(z) gating.
// ---------------------------------------------------------------------------
__global__ void scan_pass1(const float* __restrict__ dtproj,  // (ROWS, DI)
                           const float* __restrict__ dt_b,
                           const float* __restrict__ A_log,   // (DI, 16)
                           const float* __restrict__ xdbl,    // (ROWS, 64)
                           const float* __restrict__ xi32,    // (ROWS, DI)
                           float* __restrict__ yloc,          // (ROWS, DI)
                           float* __restrict__ Pch,           // (NCH, 2048, 16)
                           float* __restrict__ Hch)           // (NCH, 2048, 16)
{
  int c = (blockIdx.x & 63) * 32 + threadIdx.x;   // channel 0..2047
  int q = blockIdx.x >> 6;                        // chunk 0..NCH-1
  int b = c >> 10;
  int d = c & (DI - 1);

  float Areg[DST];
#pragma unroll
  for (int n = 0; n < DST; ++n) Areg[n] = -__expf(A_log[d * DST + n]);

  float h[DST], Pp[DST];
#pragma unroll
  for (int n = 0; n < DST; ++n) { h[n] = 0.0f; Pp[n] = 1.0f; }

  const float dbias = dt_b[d];
  const int t0 = q * CL;

  for (int tt = 0; tt < CL; ++tt) {
    size_t r = (size_t)(b * L_SZ + t0 + tt);
    float dv = dtproj[r * DI + d] + dbias;
    float delta = (dv > 20.0f) ? dv : __logf(1.0f + __expf(dv));
    float xv = xi32[r * DI + d];
    const float* bc = xdbl + r * 64 + 32;
    const float* cc = xdbl + r * 64 + 48;
    float dx = delta * xv;
    float y = 0.0f;
#pragma unroll
    for (int n = 0; n < DST; ++n) {
      float dA = __expf(delta * Areg[n]);
      Pp[n] *= dA;
      h[n] = dA * h[n] + dx * bc[n];
      y += h[n] * cc[n];
    }
    yloc[r * DI + d] = y;
  }

  size_t base = ((size_t)q * ROWS + c) * DST;
#pragma unroll
  for (int n = 0; n < DST; ++n) { Pch[base + n] = Pp[n]; Hch[base + n] = h[n]; }
}

__global__ void scan_pass2(const float* __restrict__ Pch,
                           const float* __restrict__ Hch,
                           float* __restrict__ H0ch)
{
  int c = blockIdx.x * 32 + threadIdx.x;          // 0..2047
  float h0[DST];
#pragma unroll
  for (int n = 0; n < DST; ++n) h0[n] = 0.0f;
  for (int q = 0; q < NCH; ++q) {
    size_t base = ((size_t)q * ROWS + c) * DST;
#pragma unroll
    for (int n = 0; n < DST; ++n) {
      H0ch[base + n] = h0[n];
      h0[n] = Pch[base + n] * h0[n] + Hch[base + n];
    }
  }
}

__global__ void scan_pass3(const float* __restrict__ dtproj,
                           const float* __restrict__ dt_b,
                           const float* __restrict__ A_log,
                           const float* __restrict__ xdbl,
                           const float* __restrict__ xi32,
                           const float* __restrict__ Dp,
                           const float* __restrict__ xz,      // z at col +DI
                           const float* __restrict__ yloc,
                           const float* __restrict__ H0ch,
                           _Float16* __restrict__ yg)
{
  int c = (blockIdx.x & 63) * 32 + threadIdx.x;
  int q = blockIdx.x >> 6;
  int b = c >> 10;
  int d = c & (DI - 1);
  const int t0 = q * CL;
  const float Dd = Dp[d];

  float Areg[DST], h0[DST], cumP[DST];
  const bool needCorr = (q > 0);
  if (needCorr) {
#pragma unroll
    for (int n = 0; n < DST; ++n) Areg[n] = -__expf(A_log[d * DST + n]);
    size_t base = ((size_t)q * ROWS + c) * DST;
#pragma unroll
    for (int n = 0; n < DST; ++n) { h0[n] = H0ch[base + n]; cumP[n] = 1.0f; }
  }
  const float dbias = dt_b[d];

  for (int tt = 0; tt < CL; ++tt) {
    size_t r = (size_t)(b * L_SZ + t0 + tt);
    float xv = xi32[r * DI + d];
    float corr = 0.0f;
    if (needCorr) {
      float dv = dtproj[r * DI + d] + dbias;
      float delta = (dv > 20.0f) ? dv : __logf(1.0f + __expf(dv));
      const float* cc = xdbl + r * 64 + 48;
#pragma unroll
      for (int n = 0; n < DST; ++n) {
        cumP[n] *= __expf(delta * Areg[n]);
        corr += cc[n] * cumP[n] * h0[n];
      }
    }
    float yv = yloc[r * DI + d] + corr + xv * Dd;
    float zv = xz[r * (2 * DI) + DI + d];
    float g  = zv / (1.0f + __expf(-zv));
    yg[r * DI + d] = (_Float16)(yv * g);
  }
}

// ---------------------------------------------------------------------------
// Final: out = LN(x + fused@fusion_w^T + fusion_b) * ln_g + ln_b
// ---------------------------------------------------------------------------
__global__ void bias_res_ln(const float* __restrict__ tmp,
                            const float* __restrict__ fb,
                            const float* __restrict__ x,
                            const float* __restrict__ g,
                            const float* __restrict__ beta,
                            float* __restrict__ out)
{
  __shared__ float red[256];
  int r = blockIdx.x;
  int tid = threadIdx.x;
  size_t base = (size_t)r * DM;

  float v0 = x[base + tid]       + tmp[base + tid]       + fb[tid];
  float v1 = x[base + tid + 256] + tmp[base + tid + 256] + fb[tid + 256];

  red[tid] = v0 + v1;
  __syncthreads();
  for (int o = 128; o > 0; o >>= 1) { if (tid < o) red[tid] += red[tid + o]; __syncthreads(); }
  float mu = red[0] * (1.0f / DM);
  __syncthreads();

  float d0 = v0 - mu, d1 = v1 - mu;
  red[tid] = d0 * d0 + d1 * d1;
  __syncthreads();
  for (int o = 128; o > 0; o >>= 1) { if (tid < o) red[tid] += red[tid + o]; __syncthreads(); }
  float var = red[0] * (1.0f / DM);
  float rs = rsqrtf(var + 1e-5f);

  out[base + tid]       = d0 * rs * g[tid]       + beta[tid];
  out[base + tid + 256] = d1 * rs * g[tid + 256] + beta[tid + 256];
}

// ---------------------------------------------------------------------------
// Host orchestration
// ---------------------------------------------------------------------------
extern "C" void kernel_launch(void* const* d_in, const int* in_sizes, int n_in,
                              void* d_out, int out_size, void* d_ws, size_t ws_size,
                              hipStream_t stream)
{
  (void)in_sizes; (void)n_in; (void)out_size; (void)ws_size;

  const float* x        = (const float*)d_in[0];
  const float* fusion_w = (const float*)d_in[1];
  const float* fusion_b = (const float*)d_in[2];
  const float* ln_g     = (const float*)d_in[3];
  const float* ln_b     = (const float*)d_in[4];
  const float* f_in_w   = (const float*)d_in[5];
  const float* f_conv_w = (const float*)d_in[6];
  const float* f_conv_b = (const float*)d_in[7];
  const float* f_xproj  = (const float*)d_in[8];
  const float* f_dt_w   = (const float*)d_in[9];
  const float* f_dt_b   = (const float*)d_in[10];
  const float* f_A_log  = (const float*)d_in[11];
  const float* f_D      = (const float*)d_in[12];
  const float* f_out_w  = (const float*)d_in[13];
  const float* b_in_w   = (const float*)d_in[14];
  const float* b_conv_w = (const float*)d_in[15];
  const float* b_conv_b = (const float*)d_in[16];
  const float* b_xproj  = (const float*)d_in[17];
  const float* b_dt_w   = (const float*)d_in[18];
  const float* b_dt_b   = (const float*)d_in[19];
  const float* b_A_log  = (const float*)d_in[20];
  const float* b_D      = (const float*)d_in[21];
  const float* b_out_w  = (const float*)d_in[22];

  char* p = (char*)d_ws;
  auto alloc = [&](size_t bytes) -> char* {
    char* r = p; p += (bytes + 255) & ~(size_t)255; return r;
  };

  _Float16* X16F  = (_Float16*)alloc((size_t)ROWS * DM * 2);
  _Float16* X16B  = (_Float16*)alloc((size_t)ROWS * DM * 2);
  _Float16* W_FIN = (_Float16*)alloc((size_t)2 * DI * DM * 2);
  _Float16* W_BIN = (_Float16*)alloc((size_t)2 * DI * DM * 2);
  _Float16* W_FX  = (_Float16*)alloc((size_t)64 * DI * 2);
  _Float16* W_BX  = (_Float16*)alloc((size_t)64 * DI * 2);
  _Float16* W_FDT = (_Float16*)alloc((size_t)DI * DTR * 2);
  _Float16* W_BDT = (_Float16*)alloc((size_t)DI * DTR * 2);
  _Float16* W_FO  = (_Float16*)alloc((size_t)DM * DI * 2);
  _Float16* W_BO  = (_Float16*)alloc((size_t)DM * DI * 2);
  _Float16* W_FUS = (_Float16*)alloc((size_t)DM * 2 * DM * 2);

  float* XZ_F   = (float*)alloc((size_t)ROWS * 2 * DI * 4);
  float* XZ_B   = (float*)alloc((size_t)ROWS * 2 * DI * 4);
  float* XIF32  = (float*)alloc((size_t)ROWS * DI * 4);
  float* XIB32  = (float*)alloc((size_t)ROWS * DI * 4);
  _Float16* XIF16 = (_Float16*)alloc((size_t)ROWS * DI * 2);
  _Float16* XIB16 = (_Float16*)alloc((size_t)ROWS * DI * 2);
  float* XDBL_F = (float*)alloc((size_t)ROWS * 64 * 4);
  float* XDBL_B = (float*)alloc((size_t)ROWS * 64 * 4);
  _Float16* DT16F = (_Float16*)alloc((size_t)ROWS * DTR * 2);
  _Float16* DT16B = (_Float16*)alloc((size_t)ROWS * DTR * 2);
  float* DTP_F  = (float*)alloc((size_t)ROWS * DI * 4);
  float* DTP_B  = (float*)alloc((size_t)ROWS * DI * 4);
  float* YLOC_F = (float*)alloc((size_t)ROWS * DI * 4);
  float* YLOC_B = (float*)alloc((size_t)ROWS * DI * 4);
  float* PCH_F  = (float*)alloc((size_t)NCH * ROWS * DST * 4);
  float* PCH_B  = (float*)alloc((size_t)NCH * ROWS * DST * 4);
  float* HCH_F  = (float*)alloc((size_t)NCH * ROWS * DST * 4);
  float* HCH_B  = (float*)alloc((size_t)NCH * ROWS * DST * 4);
  float* H0_F   = (float*)alloc((size_t)NCH * ROWS * DST * 4);
  float* H0_B   = (float*)alloc((size_t)NCH * ROWS * DST * 4);
  _Float16* YG16F = (_Float16*)alloc((size_t)ROWS * DI * 2);
  _Float16* YG16B = (_Float16*)alloc((size_t)ROWS * DI * 2);
  _Float16* FUSED = (_Float16*)alloc((size_t)ROWS * 2 * DM * 2);
  float* FOUT   = (float*)alloc((size_t)ROWS * DM * 4);

  auto cvt = [&](const float* src, _Float16* dst, int n) {
    cvt_f16<<<(n + 255) / 256, 256, 0, stream>>>(src, dst, n);
  };

  // 1. precision staging
  cvt(x, X16F, ROWS * DM);
  cvt_f16_flip<<<(ROWS * DM + 255) / 256, 256, 0, stream>>>(x, X16B, ROWS * DM);
  cvt(f_in_w, W_FIN, 2 * DI * DM);
  cvt(b_in_w, W_BIN, 2 * DI * DM);
  cvt(f_xproj, W_FX, 64 * DI);
  cvt(b_xproj, W_BX, 64 * DI);
  cvt(f_dt_w, W_FDT, DI * DTR);
  cvt(b_dt_w, W_BDT, DI * DTR);
  cvt(f_out_w, W_FO, DM * DI);
  cvt(b_out_w, W_BO, DM * DI);
  cvt(fusion_w, W_FUS, DM * 2 * DM);

  // 2. in_proj GEMMs
  dim3 gIn((2 * DI) / 64, ROWS / 32);
  gemm_wmma_f32<<<gIn, 32, 0, stream>>>(X16F, W_FIN, XZ_F, ROWS, 2 * DI, DM);
  gemm_wmma_f32<<<gIn, 32, 0, stream>>>(X16B, W_BIN, XZ_B, ROWS, 2 * DI, DM);

  // 3. causal depthwise conv + SiLU
  int ncv = ROWS * DI;
  conv_silu<<<(ncv + 255) / 256, 256, 0, stream>>>(XZ_F, f_conv_w, f_conv_b, XIF32, XIF16);
  conv_silu<<<(ncv + 255) / 256, 256, 0, stream>>>(XZ_B, b_conv_w, b_conv_b, XIB32, XIB16);

  // 4. x_proj GEMMs
  dim3 gX(1, ROWS / 32);
  gemm_wmma_f32<<<gX, 32, 0, stream>>>(XIF16, W_FX, XDBL_F, ROWS, 64, DI);
  gemm_wmma_f32<<<gX, 32, 0, stream>>>(XIB16, W_BX, XDBL_B, ROWS, 64, DI);

  int ndt = ROWS * DTR;
  slice_cvt_dt<<<(ndt + 255) / 256, 256, 0, stream>>>(XDBL_F, DT16F, ndt);
  slice_cvt_dt<<<(ndt + 255) / 256, 256, 0, stream>>>(XDBL_B, DT16B, ndt);

  // 5. dt_proj GEMMs
  dim3 gDt(DI / 64, ROWS / 32);
  gemm_wmma_f32<<<gDt, 32, 0, stream>>>(DT16F, W_FDT, DTP_F, ROWS, DI, DTR);
  gemm_wmma_f32<<<gDt, 32, 0, stream>>>(DT16B, W_BDT, DTP_B, ROWS, DI, DTR);

  // 6. chunk-parallel selective scan
  scan_pass1<<<64 * NCH, 32, 0, stream>>>(DTP_F, f_dt_b, f_A_log, XDBL_F, XIF32,
                                          YLOC_F, PCH_F, HCH_F);
  scan_pass1<<<64 * NCH, 32, 0, stream>>>(DTP_B, b_dt_b, b_A_log, XDBL_B, XIB32,
                                          YLOC_B, PCH_B, HCH_B);
  scan_pass2<<<64, 32, 0, stream>>>(PCH_F, HCH_F, H0_F);
  scan_pass2<<<64, 32, 0, stream>>>(PCH_B, HCH_B, H0_B);
  scan_pass3<<<64 * NCH, 32, 0, stream>>>(DTP_F, f_dt_b, f_A_log, XDBL_F, XIF32,
                                          f_D, XZ_F, YLOC_F, H0_F, YG16F);
  scan_pass3<<<64 * NCH, 32, 0, stream>>>(DTP_B, b_dt_b, b_A_log, XDBL_B, XIB32,
                                          b_D, XZ_B, YLOC_B, H0_B, YG16B);

  // 7. out_proj GEMMs -> fused f16 buffer (bwd half un-flips time)
  dim3 gOut(DM / 64, ROWS / 32);
  gemm_wmma_f16out<<<gOut, 32, 0, stream>>>(YG16F, W_FO, FUSED, ROWS, DM, DI, 2 * DM, 0, 0);
  gemm_wmma_f16out<<<gOut, 32, 0, stream>>>(YG16B, W_BO, FUSED, ROWS, DM, DI, 2 * DM, DM, 1);

  // 8. fusion GEMM
  dim3 gFus(DM / 64, ROWS / 32);
  gemm_wmma_f32<<<gFus, 32, 0, stream>>>(FUSED, W_FUS, FOUT, ROWS, DM, 2 * DM);

  // 9. bias + residual + layernorm -> d_out
  bias_res_ln<<<ROWS, 256, 0, stream>>>(FOUT, fusion_b, x, ln_g, ln_b, (float*)d_out);
}
